// CopeMultiHeadAttention_31851477467658
// MI455X (gfx1250) — matrix operand order repaired and checked by reading, not compiled, not run on hardware
//
#include <hip/hip_runtime.h>
#include <hip/hip_bf16.h>

// ---------------------------------------------------------------------------
// CoPE multi-head attention for MI455X (gfx1250), bf16 WMMA everywhere,
// TDM (tensor_load_to_lds) double-buffered staging for K and V tiles.
// B=1, S=2048, NDIMS=1024, NHEADS=16, D=64, NPOS=64.
// ---------------------------------------------------------------------------

typedef __bf16 bf16_t;
typedef __attribute__((ext_vector_type(16))) __bf16 v16bf;
typedef __attribute__((ext_vector_type(8)))  float  v8f;
typedef __attribute__((ext_vector_type(4)))  unsigned int u32x4;
typedef __attribute__((ext_vector_type(8)))  int   i32x8;
typedef __attribute__((ext_vector_type(4)))  int   i32x4;

#define WMMA_BF16(a, b, c) \
  __builtin_amdgcn_wmma_f32_16x16x32_bf16(false, (a), false, (b), (short)0, (c), false, false)

__device__ __forceinline__ int lane_id() { return (int)(threadIdx.x & 31); }

// ----- Fragment loaders: contiguous k order (k = 16*half + s), same for A and B --------------------
__device__ __forceinline__ v16bf load_frag_a(const bf16_t* __restrict__ A, int lda) {
  const int lane = lane_id();
  const int m = lane & 15, half = lane >> 4;
  v16bf f;
#pragma unroll
  for (int j = 0; j < 8; ++j) {
    const int k = (half << 4) + (j << 1);
    f[2 * j]     = A[m * lda + k];
    f[2 * j + 1] = A[m * lda + k + 1];
  }
  return f;
}

__device__ __forceinline__ v16bf load_frag_a_f32(const float* __restrict__ A, int lda) {
  const int lane = lane_id();
  const int m = lane & 15, half = lane >> 4;
  v16bf f;
#pragma unroll
  for (int j = 0; j < 8; ++j) {
    const int k = (half << 4) + (j << 1);
    f[2 * j]     = (bf16_t)A[m * lda + k];
    f[2 * j + 1] = (bf16_t)A[m * lda + k + 1];
  }
  return f;
}

__device__ __forceinline__ v16bf load_frag_b(const bf16_t* __restrict__ B, int ldb) {
  const int lane = lane_id();
  const int n = lane & 15, half = lane >> 4;
  v16bf f;
#pragma unroll
  for (int j = 0; j < 8; ++j) {
    const int k = (half << 4) + (j << 1);
    f[2 * j]     = B[k * ldb + n];
    f[2 * j + 1] = B[(k + 1) * ldb + n];
  }
  return f;
}

__device__ __forceinline__ v16bf load_frag_bt(const bf16_t* __restrict__ B, int ldb) {
  const int lane = lane_id();
  const int n = lane & 15, half = lane >> 4;
  v16bf f;
#pragma unroll
  for (int j = 0; j < 8; ++j) {
    const int k = (half << 4) + (j << 1);
    f[2 * j]     = B[n * ldb + k];
    f[2 * j + 1] = B[n * ldb + k + 1];
  }
  return f;
}

__device__ __forceinline__ v16bf load_frag_b_f32(const float* __restrict__ B, int ldb) {
  const int lane = lane_id();
  const int n = lane & 15, half = lane >> 4;
  v16bf f;
#pragma unroll
  for (int j = 0; j < 8; ++j) {
    const int k = (half << 4) + (j << 1);
    f[2 * j]     = (bf16_t)B[k * ldb + n];
    f[2 * j + 1] = (bf16_t)B[(k + 1) * ldb + n];
  }
  return f;
}

// ---------------------------------------------------------------------------
// Tensor Data Mover: 2D tile load Global->LDS via hand-packed D#.
// ---------------------------------------------------------------------------
#if __has_builtin(__builtin_amdgcn_tensor_load_to_lds)
#define HAVE_TDM 1
#pragma message("CDNA5: tensor_load_to_lds builtin AVAILABLE - TDM staging enabled")
__device__ __forceinline__ void tdm_load_2d(void* lds_ptr, const void* gptr,
                                            unsigned dim0, unsigned dim1,
                                            unsigned tile0, unsigned tile1,
                                            unsigned long long stride0,
                                            unsigned data_size_code /*0=1B,1=2B,2=4B*/) {
  const unsigned long long ga = (unsigned long long)(size_t)gptr;
  const unsigned lds_addr = (unsigned)(size_t)lds_ptr;  // LDS aperture: addr[31:0]
  u32x4 g0;
  g0[0] = 1u;                                                 // count=1 (valid user D#)
  g0[1] = lds_addr;                                           // bits [63:32]
  g0[2] = (unsigned)(ga & 0xFFFFFFFFu);                       // global_addr[31:0]
  g0[3] = (unsigned)((ga >> 32) & 0x1FFFFFFu) | (2u << 30);   // addr[56:32] | type=2
  i32x8 g1;
  g1[0] = (int)(data_size_code << 16);                        // wg_mask=0 | data_size
  g1[1] = (int)((dim0 & 0xFFFFu) << 16);                      // tensor_dim0[15:0]
  g1[2] = (int)(((dim0 >> 16) & 0xFFFFu) | ((dim1 & 0xFFFFu) << 16));
  g1[3] = (int)(((dim1 >> 16) & 0xFFFFu) | ((tile0 & 0xFFFFu) << 16));
  g1[4] = (int)(tile1 & 0xFFFFu);                             // tile_dim1 | tile_dim2=0
  g1[5] = (int)(unsigned)(stride0 & 0xFFFFFFFFull);           // tensor_dim0_stride
  g1[6] = (int)((stride0 >> 32) & 0xFFFFull);
  g1[7] = 0;                                                  // tensor_dim1_stride=0 (2D)
  i32x4 z4 = {0, 0, 0, 0};
#if __clang_major__ >= 23
  i32x8 z8 = {0, 0, 0, 0, 0, 0, 0, 0};
  __builtin_amdgcn_tensor_load_to_lds(g0, g1, z4, z4, z8, 0);
#else
  __builtin_amdgcn_tensor_load_to_lds(g0, g1, z4, z4, 0);
#endif
}
#else
#define HAVE_TDM 0
#pragma message("CDNA5: tensor_load_to_lds builtin NOT available - fallback loads")
#endif

// ---------------------------------------------------------------------------
// fp32 -> bf16 conversion
// ---------------------------------------------------------------------------
__global__ void f32_to_bf16_kernel(const float* __restrict__ in, bf16_t* __restrict__ out, int n) {
  int i = blockIdx.x * blockDim.x + threadIdx.x;
  if (i < n) out[i] = (bf16_t)in[i];
}

// ---------------------------------------------------------------------------
// C[M x N] = A[M x K](bf16) * W[N x K]^T (bf16) + bias.  4 waves/block, each
// wave owns a 16x64 slice of a 64x64 block tile.
// ---------------------------------------------------------------------------
template <bool OUT_F32>
__global__ __launch_bounds__(128)
void gemm_xWT_bias(const bf16_t* __restrict__ A, const bf16_t* __restrict__ W,
                   const float* __restrict__ bias, void* __restrict__ Cout,
                   int M, int N, int K) {
  const int wave = threadIdx.x >> 5;
  const int m0 = blockIdx.y * 64 + wave * 16;
  const int n0 = blockIdx.x * 64;

  v8f acc[4] = {};
  const bf16_t* Arow = A + (size_t)m0 * K;

  for (int k0 = 0; k0 < K; k0 += 32) {
    if (k0 + 32 < K) {  // gfx1250 global_prefetch path
      __builtin_prefetch((const void*)(Arow + k0 + 32), 0, 0);
      __builtin_prefetch((const void*)(W + (size_t)n0 * K + k0 + 32), 0, 0);
    }
    v16bf a = load_frag_a(Arow + k0, K);
#pragma unroll
    for (int t = 0; t < 4; ++t) {
      v16bf b = load_frag_bt(W + (size_t)(n0 + 16 * t) * K + k0, K);
      acc[t] = WMMA_BF16(a, b, acc[t]);
    }
  }

  const int lane = lane_id();
  const int n = lane & 15, half = lane >> 4;
#pragma unroll
  for (int t = 0; t < 4; ++t) {
#pragma unroll
    for (int v = 0; v < 8; ++v) {
      const int row = m0 + v + 8 * half;
      const int col = n0 + 16 * t + n;
      const float val = acc[t][v] + bias[col];
      if constexpr (OUT_F32)
        ((float*)Cout)[(size_t)row * N + col] = val;
      else
        ((bf16_t*)Cout)[(size_t)row * N + col] = (bf16_t)val;
    }
  }
}

// ---------------------------------------------------------------------------
// Fused CoPE attention: 16 waves per (head, 16-query tile) block.
//   phase 0: TDM-stage Q tile (16x64 bf16) into LDS
//   phase 1: logits = (Q K^T)/8 -> LDS [16][S]; per-wave TDM double-buffered
//            K tiles (16x64) through private LDS staging slots
//            li = Q @ pos_emb  -> LDS [16][64]   (wave 0)
//   phase 2: CoPE bias + softmax, one wave per query row, lane-parallel
//            suffix-scan of sigmoid gates (chunk = 64 keys/lane)
//   phase 3: O = scores @ V; 4 waves, one 16-wide N-tile each, per-wave TDM
//            double-buffered V tiles (32x16)
// ---------------------------------------------------------------------------
__global__ __launch_bounds__(512)
void cope_attn_kernel(const bf16_t* __restrict__ Q, const bf16_t* __restrict__ Kmat,
                      const bf16_t* __restrict__ V, const float* __restrict__ pos_emb,
                      bf16_t* __restrict__ O) {
  constexpr int S = 2048, ND = 1024, D = 64, NPOS = 64;
  constexpr int LROW = S + 16;   // padded LDS row stride (floats)
  constexpr int CHUNK = S / 32;  // keys per lane in the scan phase
  constexpr int KTILES = S / 16; // 128 k-tiles
  constexpr int NT = KTILES / 16;// 8 k-tiles per wave in phase 1

  extern __shared__ float smem[];
  float*  Lg   = smem;                     // [16][LROW] logits -> scores
  float*  Li   = Lg + 16 * LROW;           // [16][NPOS] interpolation table
  float*  Rsum = Li + 16 * NPOS;           // [16] softmax denominators
  bf16_t* Qs   = (bf16_t*)(Rsum + 16);     // [16][64] staged Q tile
  bf16_t* Ks   = Qs + 16 * D;              // [16 waves][2][16*64] staging (64KB)

  const int h    = blockIdx.y;
  const int q0   = blockIdx.x * 16;
  const int wave = threadIdx.x >> 5;
  const int lane = lane_id();
  const int n = lane & 15, half = lane >> 4;

  // ---- phase 0: stage Q tile into LDS ------------------------------------
  const bf16_t* Qh = Q + (size_t)q0 * ND + h * D;
#if HAVE_TDM
  if (wave == 0) {
    tdm_load_2d((void*)Qs, (const void*)Qh,
                /*dim0=*/D, /*dim1=*/16, /*tile0=*/D, /*tile1=*/16,
                /*stride0=*/(unsigned long long)ND, /*dsize=*/1u);
    __builtin_amdgcn_s_wait_tensorcnt(0);
  }
#else
  for (int i = threadIdx.x; i < 16 * D; i += 512)
    Qs[i] = Qh[(size_t)(i >> 6) * ND + (i & 63)];
#endif
  __syncthreads();

  // Q tile fragments from LDS (row stride 64).
  const v16bf a0 = load_frag_a(Qs, D);
  const v16bf a1 = load_frag_a(Qs + 32, D);

  // ---- phase 1: logits tiles, 8 per wave ----------------------------------
  const float scale = 0.125f;  // 1/sqrt(64)
#if HAVE_TDM
  {
    bf16_t* myK = Ks + wave * (2 * 16 * D);  // private double buffer
    tdm_load_2d((void*)myK, (const void*)(Kmat + (size_t)(wave * 16) * ND + h * D),
                D, 16, D, 16, (unsigned long long)ND, 1u);
    for (int i = 0; i < NT; ++i) {
      bf16_t* cur = myK + (i & 1) * (16 * D);
      bf16_t* nxt = myK + ((i & 1) ^ 1) * (16 * D);
      if (i + 1 < NT) {
        const int ktn = wave + 16 * (i + 1);
        tdm_load_2d((void*)nxt, (const void*)(Kmat + (size_t)(ktn * 16) * ND + h * D),
                    D, 16, D, 16, (unsigned long long)ND, 1u);
        __builtin_amdgcn_s_wait_tensorcnt(1);  // oldest (cur) done, next in flight
      } else {
        __builtin_amdgcn_s_wait_tensorcnt(0);
      }
      const int kt = wave + 16 * i;
      v16bf b0 = load_frag_bt(cur, D);        // staged tile: [key][dim], stride 64
      v16bf b1 = load_frag_bt(cur + 32, D);
      v8f c = {};
      c = WMMA_BF16(a0, b0, c);
      c = WMMA_BF16(a1, b1, c);
#pragma unroll
      for (int v = 0; v < 8; ++v)
        Lg[(v + 8 * half) * LROW + kt * 16 + n] = c[v] * scale;
    }
  }
#else
  for (int kt = wave; kt < KTILES; kt += 16) {
    const bf16_t* Kp = Kmat + (size_t)(kt * 16) * ND + h * D;
    v16bf b0 = load_frag_bt(Kp, ND);
    v16bf b1 = load_frag_bt(Kp + 32, ND);
    v8f c = {};
    c = WMMA_BF16(a0, b0, c);
    c = WMMA_BF16(a1, b1, c);
#pragma unroll
    for (int v = 0; v < 8; ++v)
      Lg[(v + 8 * half) * LROW + kt * 16 + n] = c[v] * scale;
  }
#endif

  // li = Q @ pos_emb (16 x 64), wave 0 only
  if (wave == 0) {
#pragma unroll
    for (int t = 0; t < 4; ++t) {
      v16bf b0 = load_frag_b_f32(pos_emb + t * 16, NPOS);
      v16bf b1 = load_frag_b_f32(pos_emb + 32 * NPOS + t * 16, NPOS);
      v8f c = {};
      c = WMMA_BF16(a0, b0, c);
      c = WMMA_BF16(a1, b1, c);
#pragma unroll
      for (int v = 0; v < 8; ++v)
        Li[(v + 8 * half) * NPOS + t * 16 + n] = c[v];
    }
  }
  __syncthreads();

  // ---- phase 2: CoPE + softmax; wave w owns query row w -------------------
  {
    float* lg = Lg + wave * LROW;
    const float* li = Li + wave * NPOS;
    const int kbase = lane * CHUNK;

    // (a) per-lane chunk gate-sum
    float cs = 0.f;
    for (int k = kbase; k < kbase + CHUNK; ++k)
      cs += 1.f / (1.f + __expf(-lg[k]));

    // (b) inclusive suffix-sum across lanes, then exclusive
    float incl = cs;
#pragma unroll
    for (int off = 1; off < 32; off <<= 1) {
      float t = __shfl_down(incl, off, 32);
      if (lane + off < 32) incl += t;
    }
    float run = incl - cs;  // gates for all keys after this chunk

    // (c) serial reverse scan inside own chunk: CoPE bias + adj logits + max
    float mx = -3.0e38f;
    for (int k = kbase + CHUNK - 1; k >= kbase; --k) {
      const float l = lg[k];
      run += 1.f / (1.f + __expf(-l));
      float pos = fminf(run, (float)(NPOS - 1));
      const float pf = floorf(pos);
      const float pc = ceilf(pos);
      const float w = pos - pf;
      const float adj = l + li[(int)pc] * w + li[(int)pf] * (1.f - w);
      lg[k] = adj;
      mx = fmaxf(mx, adj);
    }
#pragma unroll
    for (int off = 1; off < 32; off <<= 1)
      mx = fmaxf(mx, __shfl_xor(mx, off, 32));

    // (d) unnormalized softmax numerators + row sum
    float s = 0.f;
    for (int k = kbase; k < kbase + CHUNK; ++k) {
      const float e = __expf(lg[k] - mx);
      lg[k] = e;
      s += e;
    }
#pragma unroll
    for (int off = 1; off < 32; off <<= 1)
      s += __shfl_xor(s, off, 32);
    if (lane == 0) Rsum[wave] = s;
  }
  __syncthreads();

  // ---- phase 3: O = scores @ V; wave t owns N-tile t ----------------------
  if (wave < 4) {
    const int t = wave;
    v8f o = {};
#if HAVE_TDM
    bf16_t* myV = Ks + wave * (2 * 32 * 16);  // private double buffer (Ks reused)
    tdm_load_2d((void*)myV, (const void*)(V + h * D + t * 16),
                /*dim0=*/16, /*dim1=*/32, /*tile0=*/16, /*tile1=*/32,
                (unsigned long long)ND, 1u);
    for (int i = 0; i < S / 32; ++i) {
      bf16_t* cur = myV + (i & 1) * (32 * 16);
      bf16_t* nxt = myV + ((i & 1) ^ 1) * (32 * 16);
      if (i + 1 < S / 32) {
        tdm_load_2d((void*)nxt,
                    (const void*)(V + (size_t)((i + 1) * 32) * ND + h * D + t * 16),
                    16, 32, 16, 32, (unsigned long long)ND, 1u);
        __builtin_amdgcn_s_wait_tensorcnt(1);
      } else {
        __builtin_amdgcn_s_wait_tensorcnt(0);
      }
      v16bf p = load_frag_a_f32(Lg + i * 32, LROW);  // P tile from LDS
      v16bf b = load_frag_b(cur, 16);                // staged V: [key][dim], stride 16
      o = WMMA_BF16(p, b, o);
    }
#else
    for (int k0 = 0; k0 < S; k0 += 32) {
      v16bf p = load_frag_a_f32(Lg + k0, LROW);
      v16bf b = load_frag_b(V + (size_t)k0 * ND + h * D + t * 16, ND);
      o = WMMA_BF16(p, b, o);
    }
#endif
#pragma unroll
    for (int v = 0; v < 8; ++v) {
      const int m = v + 8 * half;
      O[(size_t)(q0 + m) * ND + h * D + t * 16 + n] = (bf16_t)(o[v] / Rsum[m]);
    }
  }
}

// ---------------------------------------------------------------------------
// Host-side launcher
// ---------------------------------------------------------------------------
extern "C" void kernel_launch(void* const* d_in, const int* in_sizes, int n_in,
                              void* d_out, int out_size, void* d_ws, size_t ws_size,
                              hipStream_t stream) {
  (void)in_sizes; (void)n_in; (void)out_size; (void)ws_size;
  constexpr int S = 2048, ND = 1024, NH = 16, NPOS = 64;

  const float* q      = (const float*)d_in[0];
  const float* k      = (const float*)d_in[1];
  const float* v      = (const float*)d_in[2];
  const float* Wq_w   = (const float*)d_in[3];
  const float* Wq_b   = (const float*)d_in[4];
  const float* Wk_w   = (const float*)d_in[5];
  const float* Wk_b   = (const float*)d_in[6];
  const float* Wv_w   = (const float*)d_in[7];
  const float* Wv_b   = (const float*)d_in[8];
  const float* Wo_w   = (const float*)d_in[9];
  const float* Wo_b   = (const float*)d_in[10];
  const float* posemb = (const float*)d_in[11];
  float* out = (float*)d_out;

  // Workspace layout (bf16): x_{q,k,v}, W_{q,k,v,o}, Q, K, V, attn-out
  const size_t xn = (size_t)S * ND;
  const size_t wn = (size_t)ND * ND;
  bf16_t* ws = (bf16_t*)d_ws;
  bf16_t* xq = ws;
  bf16_t* xk = xq + xn;
  bf16_t* xv = xk + xn;
  bf16_t* wq = xv + xn;
  bf16_t* wk = wq + wn;
  bf16_t* wv = wk + wn;
  bf16_t* wo = wv + wn;
  bf16_t* Qb = wo + wn;
  bf16_t* Kb = Qb + xn;
  bf16_t* Vb = Kb + xn;
  bf16_t* AO = Vb + xn;

  auto cvt = [&](const float* src, bf16_t* dst, size_t n) {
    const int thr = 256;
    f32_to_bf16_kernel<<<(unsigned)((n + thr - 1) / thr), thr, 0, stream>>>(src, dst, (int)n);
  };
  cvt(q, xq, xn); cvt(k, xk, xn); cvt(v, xv, xn);
  cvt(Wq_w, wq, wn); cvt(Wk_w, wk, wn); cvt(Wv_w, wv, wn); cvt(Wo_w, wo, wn);

  // Projections: Q/K/V = x @ W^T + b  (bf16 out)
  dim3 ggrid(ND / 64, S / 64);
  gemm_xWT_bias<false><<<ggrid, 128, 0, stream>>>(xq, wq, Wq_b, Qb, S, ND, ND);
  gemm_xWT_bias<false><<<ggrid, 128, 0, stream>>>(xk, wk, Wk_b, Kb, S, ND, ND);
  gemm_xWT_bias<false><<<ggrid, 128, 0, stream>>>(xv, wv, Wv_b, Vb, S, ND, ND);

  // Fused CoPE attention: 128 q-tiles x 16 heads, 16 waves each, big dynamic LDS.
  // Layout must match device: Lg + Li + Rsum + Qs + Ks(64KB staging).
  const size_t smem_bytes =
      (size_t)(16 * (S + 16) + 16 * NPOS + 16) * sizeof(float) +
      (size_t)16 * 64 * sizeof(bf16_t) +              // Qs
      (size_t)16 * 2 * 16 * 64 * sizeof(bf16_t);      // Ks staging
  hipFuncSetAttribute(reinterpret_cast<const void*>(cope_attn_kernel),
                      hipFuncAttributeMaxDynamicSharedMemorySize, (int)smem_bytes);
  cope_attn_kernel<<<dim3(S / 16, NH), 512, smem_bytes, stream>>>(Qb, Kb, Vb, posemb, AO);

  // Output projection: out = AO @ Wo^T + b (fp32 out)
  gemm_xWT_bias<true><<<ggrid, 128, 0, stream>>>(AO, wo, Wo_b, out, S, ND, ND);
}
